// MixtureLayer_17025250361619
// MI455X (gfx1250) — matrix operand order repaired and checked by reading, not compiled, not run on hardware
//
#include <hip/hip_runtime.h>

// ---------------- types ----------------
typedef __bf16 bf16x16 __attribute__((ext_vector_type(16)));
typedef float  f32x8   __attribute__((ext_vector_type(8)));

union Frag { bf16x16 v; uint4 q[2]; };

#define D_MODEL 1024
#define HIDDEN  4096
#define NEXP    8
#define NGROUPS 128
#define GROUP   128
#define NTOK    16384           // 8*2048
#define NSLOT   32              // capacity buffer (slots 0..30 used)

#define AP 1032                 // LDS A row stride (elements), 2064B = 129*16, +4 banks/row
#define HP 2056                 // LDS hidden row stride (elements), 4112B, +4 banks/row

__device__ __forceinline__ unsigned short f2bf(float f) {
  unsigned int u = __float_as_uint(f);
  unsigned int r = u + 0x7FFFu + ((u >> 16) & 1u);
  return (unsigned short)(r >> 16);
}
__device__ __forceinline__ float bf2f(unsigned int h) {
  return __uint_as_float(h << 16);
}
// branch-free tanh-GELU: tanh via raw v_exp_f32 (no libm, no exec divergence)
__device__ __forceinline__ float gelu_tanh(float x) {
  const float k = 0.7978845608028654f;
  float y = k * (x + 0.044715f * x * x * x);
  // exp(-2|y|) = exp2(-2*log2(e)*|y|)
  float t = __builtin_amdgcn_exp2f(-2.8853900817779268f * fabsf(y));
  float th = (1.0f - t) * __builtin_amdgcn_rcpf(1.0f + t);  // tanh(|y|)
  th = copysignf(th, y);
  return 0.5f * x * (1.0f + th);
}
__device__ __forceinline__ f32x8 wmma_bf16(const Frag& a, const Frag& b, f32x8 c) {
  return __builtin_amdgcn_wmma_f32_16x16x32_bf16(false, a.v, false, b.v, (short)0, c,
                                                 false, false);
}

// ---------------- fp32 -> bf16 pair convert (x) ----------------
__global__ __launch_bounds__(256) void convx_kernel(const float* __restrict__ src,
                                                    unsigned int* __restrict__ dst,
                                                    long long n2) {
  long long i = (long long)blockIdx.x * blockDim.x + threadIdx.x;
  if (i < n2) {
    float a = src[2 * i], b = src[2 * i + 1];
    dst[i] = (unsigned int)f2bf(a) | ((unsigned int)f2bf(b) << 16);
  }
}

// ---------------- fp32 -> bf16 transpose (weights) ----------------
// src[b][r][c] (f32)  ->  dst[b][c][r] (bf16)
__global__ __launch_bounds__(256) void trans_kernel(const float* __restrict__ src,
                                                    unsigned short* __restrict__ dst,
                                                    int R, int C) {
  __shared__ float tile[32][33];
  int b = blockIdx.z;
  long long base = (long long)b * R * C;
  int c0 = blockIdx.x * 32, r0 = blockIdx.y * 32;
  int tx = threadIdx.x & 31, ty = threadIdx.x >> 5;
#pragma unroll
  for (int i = 0; i < 32; i += 8)
    tile[ty + i][tx] = src[base + (long long)(r0 + ty + i) * C + c0 + tx];
  __syncthreads();
#pragma unroll
  for (int i = 0; i < 32; i += 8)
    dst[base + (long long)(c0 + ty + i) * R + r0 + tx] = f2bf(tile[tx][ty + i]);
}

// ---------------- gating: softmax top-2, per-(k,e) capacity cumsum ----------------
__global__ __launch_bounds__(128) void gate_kernel(const float* __restrict__ x,
                                                   const float* __restrict__ gw,
                                                   const float* __restrict__ gb,
                                                   int* __restrict__ disp,
                                                   int* __restrict__ tokE,
                                                   int* __restrict__ tokC,
                                                   float* __restrict__ tokG) {
  __shared__ float sgw[D_MODEL * NEXP];
  __shared__ int   sTopE[GROUP][2];
  int g = blockIdx.x, tid = threadIdx.x;
  for (int i = tid; i < D_MODEL * NEXP; i += 128) sgw[i] = gw[i];
  for (int i = tid; i < NEXP * NSLOT * 2; i += 128) disp[g * (NEXP * NSLOT * 2) + i] = -1;
  __syncthreads();

  int s = tid;
  float acc[NEXP];
#pragma unroll
  for (int e = 0; e < NEXP; ++e) acc[e] = gb[e];
  const float* xr = x + (long long)(g * GROUP + s) * D_MODEL;
  for (int d = 0; d < D_MODEL; ++d) {
    float xv = xr[d];
#pragma unroll
    for (int e = 0; e < NEXP; ++e) acc[e] += xv * sgw[d * NEXP + e];
  }
  // softmax
  float m = acc[0];
#pragma unroll
  for (int e = 1; e < NEXP; ++e) m = fmaxf(m, acc[e]);
  float sum = 0.f;
#pragma unroll
  for (int e = 0; e < NEXP; ++e) {
    acc[e] = __builtin_amdgcn_exp2f(1.4426950408889634f * (acc[e] - m));
    sum += acc[e];
  }
  float inv = 1.0f / sum;
#pragma unroll
  for (int e = 0; e < NEXP; ++e) acc[e] *= inv;
  // top-2 (first index wins ties, like jax top_k)
  int i0 = 0;
#pragma unroll
  for (int e = 1; e < NEXP; ++e) if (acc[e] > acc[i0]) i0 = e;
  int i1 = (i0 == 0) ? 1 : 0;
#pragma unroll
  for (int e = 0; e < NEXP; ++e) if (e != i0 && acc[e] > acc[i1]) i1 = e;

  sTopE[s][0] = i0; sTopE[s][1] = i1;
  long long tb = (long long)(g * GROUP + s) * 2;
  tokE[tb + 0] = i0;       tokE[tb + 1] = i1;
  tokG[tb + 0] = acc[i0];  tokG[tb + 1] = acc[i1];
  __syncthreads();

  if (tid < NEXP) {
    int e = tid;
    int cnt[2] = {0, 0};
    for (int s2 = 0; s2 < GROUP; ++s2)
      for (int k = 0; k < 2; ++k)
        if (sTopE[s2][k] == e) {
          int pos = ++cnt[k];                        // cumsum incl. self -> pos>=1
          int c = (pos <= NSLOT - 1) ? pos - 1 : -1; // slot 0 dropped by ref
          if (c >= 0) disp[((g * NEXP + e) * NSLOT + c) * 2 + k] = s2;
          tokC[(long long)(g * GROUP + s2) * 2 + k] = c;
        }
  }
}

// ---------------- fragment loaders ----------------
// A (16x32 bf16) from LDS row-major (stride in elements). ISA A layout:
// lane = m + 16h; per-lane Ks: {8h..8h+7} U {16+8h..16+8h+7} -> two b128 loads.
__device__ __forceinline__ void loadA(Frag& f, const unsigned short* base, int stride,
                                      int m0, int k0, int lr, int lh) {
  const unsigned short* p = base + (m0 + lr) * stride + k0 + 8 * lh;
  f.q[0] = *(const uint4*)p;
  f.q[1] = *(const uint4*)(p + 16);
}
// B (32x16 bf16) from global, B stored row-major as [n][k] (transposed weight).
// lane = n + 16h; per-lane Ks: {8h..8h+7} U {16+8h..16+8h+7} -> two b128 loads.
__device__ __forceinline__ void loadB(Frag& f, const unsigned short* base, int stride,
                                      int n, int k0, int lh) {
  const unsigned short* p = base + (long long)n * stride + k0 + 8 * lh;
  f.q[0] = *(const uint4*)p;
  f.q[1] = *(const uint4*)(p + 16);
}

#define ZERO8 {0.f, 0.f, 0.f, 0.f, 0.f, 0.f, 0.f, 0.f}

// ---------------- per-(g,e) routed expert FFN ----------------
__global__ __launch_bounds__(256) void expert_kernel(const unsigned short* __restrict__ xb,
                                                     const unsigned short* __restrict__ kb,
                                                     const unsigned short* __restrict__ vb,
                                                     const int* __restrict__ disp,
                                                     unsigned short* __restrict__ eo) {
  extern __shared__ char smem[];
  unsigned short* A  = (unsigned short*)smem;                       // 32*AP
  unsigned short* Hd = (unsigned short*)(smem + 32 * AP * 2);       // 32*HP
  int* slotTok = (int*)(smem + 32 * AP * 2 + 32 * HP * 2);          // 64 ints

  int e = blockIdx.x >> 7;        // e-major: 128 consecutive blocks share expert (L2)
  int g = blockIdx.x & 127;
  int tid = threadIdx.x, lane = tid & 31, wave = tid >> 5;
  int lr = lane & 15, lh = lane >> 4;

  if (tid < 64) slotTok[tid] = disp[(long long)(g * NEXP + e) * NSLOT * 2 + tid];
  __syncthreads();

  // gather A rows: sum of (up to 2) routed tokens per slot (k-lane collisions)
  unsigned int* Au = (unsigned int*)A;
  const unsigned int* xbu = (const unsigned int*)xb;
  for (int u = tid; u < 32 * (D_MODEL / 2); u += 256) {
    int row = u >> 9, col = u & 511;
    int t0 = slotTok[row * 2 + 0], t1 = slotTok[row * 2 + 1];
    unsigned int a0 = (t0 >= 0) ? xbu[(long long)(g * GROUP + t0) * (D_MODEL / 2) + col] : 0u;
    unsigned int a1 = (t1 >= 0) ? xbu[(long long)(g * GROUP + t1) * (D_MODEL / 2) + col] : 0u;
    float lo = bf2f(a0 & 0xffffu) + bf2f(a1 & 0xffffu);
    float hi = bf2f(a0 >> 16) + bf2f(a1 >> 16);
    Au[row * (AP / 2) + col] = (unsigned int)f2bf(lo) | ((unsigned int)f2bf(hi) << 16);
  }
  __syncthreads();

  const unsigned short* kbE = kb + (long long)e * HIDDEN * D_MODEL;  // [h][d]
  const unsigned short* vbE = vb + (long long)e * D_MODEL * HIDDEN;  // [d][h]

  f32x8 oacc[2][8];
#pragma unroll
  for (int mt = 0; mt < 2; ++mt)
#pragma unroll
    for (int nt = 0; nt < 8; ++nt) oacc[mt][nt] = (f32x8)ZERO8;

  for (int ch = 0; ch < 2; ++ch) {
    int hbase = ch * 2048;
    // ---- GEMM1: hidden = A(32x1024) @ K(1024 x chunk2048), per-wave 256 cols
    int n0w = wave * 256;
#pragma unroll 1
    for (int nb = 0; nb < 4; ++nb) {
      int n0 = n0w + nb * 64;
      f32x8 hacc[2][4];
#pragma unroll
      for (int mt = 0; mt < 2; ++mt)
#pragma unroll
        for (int nt = 0; nt < 4; ++nt) hacc[mt][nt] = (f32x8)ZERO8;
#pragma unroll 1
      for (int kt = 0; kt < 32; ++kt) {
        int k0 = kt * 32;
        Frag a0, a1;
        loadA(a0, A, AP, 0, k0, lr, lh);
        loadA(a1, A, AP, 16, k0, lr, lh);
#pragma unroll
        for (int nt = 0; nt < 4; ++nt) {
          Frag b;
          loadB(b, kbE, D_MODEL, hbase + n0 + nt * 16 + lr, k0, lh);
          hacc[0][nt] = wmma_bf16(a0, b, hacc[0][nt]);
          hacc[1][nt] = wmma_bf16(a1, b, hacc[1][nt]);
        }
      }
      // GELU -> bf16 hidden tile in LDS
#pragma unroll
      for (int mt = 0; mt < 2; ++mt)
#pragma unroll
        for (int nt = 0; nt < 4; ++nt)
#pragma unroll
          for (int r = 0; r < 8; ++r) {
            int mm = mt * 16 + r + 8 * lh;
            int cc = n0 + nt * 16 + lr;
            Hd[mm * HP + cc] = f2bf(gelu_tanh(hacc[mt][nt][r]));
          }
    }
    __syncthreads();
    // ---- GEMM2: out += hidden(32 x 2048) @ V(2048 x 1024), per-wave 128 out cols
    int d0 = wave * 128;
#pragma unroll 1
    for (int kt = 0; kt < 64; ++kt) {
      int k0 = kt * 32;
      Frag a0, a1;
      loadA(a0, Hd, HP, 0, k0, lr, lh);
      loadA(a1, Hd, HP, 16, k0, lr, lh);
#pragma unroll
      for (int nt = 0; nt < 8; ++nt) {
        Frag b;
        loadB(b, vbE, HIDDEN, d0 + nt * 16 + lr, hbase + k0, lh);
        oacc[0][nt] = wmma_bf16(a0, b, oacc[0][nt]);
        oacc[1][nt] = wmma_bf16(a1, b, oacc[1][nt]);
      }
    }
    __syncthreads();
  }

  unsigned short* eoB = eo + (long long)(g * NEXP + e) * NSLOT * D_MODEL;
#pragma unroll
  for (int mt = 0; mt < 2; ++mt)
#pragma unroll
    for (int nt = 0; nt < 8; ++nt)
#pragma unroll
      for (int r = 0; r < 8; ++r) {
        int mm = mt * 16 + r + 8 * lh;
        int dd = wave * 128 + nt * 16 + lr;
        eoB[mm * D_MODEL + dd] = f2bf(oacc[mt][nt][r]);
      }
}

// ---------------- shared expert + final combine ----------------
__global__ __launch_bounds__(256) void shared_kernel(const unsigned short* __restrict__ xb,
                                                     const unsigned short* __restrict__ skb,
                                                     const unsigned short* __restrict__ svb,
                                                     const int* __restrict__ tokE,
                                                     const int* __restrict__ tokC,
                                                     const float* __restrict__ tokG,
                                                     const unsigned short* __restrict__ eo,
                                                     float* __restrict__ out) {
  extern __shared__ char smem[];
  unsigned short* A  = (unsigned short*)smem;
  unsigned short* Hd = (unsigned short*)(smem + 32 * AP * 2);
  int*   mE = (int*)(smem + 32 * AP * 2 + 32 * HP * 2);   // 64 ints
  int*   mC = mE + 64;                                    // 64 ints
  float* mG = (float*)(mC + 64);                          // 64 floats

  long long row0 = (long long)blockIdx.x * 32;
  int g = (int)(row0 >> 7);
  int tid = threadIdx.x, lane = tid & 31, wave = tid >> 5;
  int lr = lane & 15, lh = lane >> 4;

  if (tid < 64) {
    mE[tid] = tokE[row0 * 2 + tid];
    mC[tid] = tokC[row0 * 2 + tid];
    mG[tid] = tokG[row0 * 2 + tid];
  }
  unsigned int* Au = (unsigned int*)A;
  const unsigned int* xbu = (const unsigned int*)xb;
  for (int u = tid; u < 32 * (D_MODEL / 2); u += 256) {
    int row = u >> 9, col = u & 511;
    Au[row * (AP / 2) + col] = xbu[(row0 + row) * (D_MODEL / 2) + col];
  }
  __syncthreads();

  f32x8 oacc[2][8];
#pragma unroll
  for (int mt = 0; mt < 2; ++mt)
#pragma unroll
    for (int nt = 0; nt < 8; ++nt) oacc[mt][nt] = (f32x8)ZERO8;

  for (int ch = 0; ch < 2; ++ch) {
    int hbase = ch * 2048;
    int n0w = wave * 256;
#pragma unroll 1
    for (int nb = 0; nb < 4; ++nb) {
      int n0 = n0w + nb * 64;
      f32x8 hacc[2][4];
#pragma unroll
      for (int mt = 0; mt < 2; ++mt)
#pragma unroll
        for (int nt = 0; nt < 4; ++nt) hacc[mt][nt] = (f32x8)ZERO8;
#pragma unroll 1
      for (int kt = 0; kt < 32; ++kt) {
        int k0 = kt * 32;
        Frag a0, a1;
        loadA(a0, A, AP, 0, k0, lr, lh);
        loadA(a1, A, AP, 16, k0, lr, lh);
#pragma unroll
        for (int nt = 0; nt < 4; ++nt) {
          Frag b;
          loadB(b, skb, D_MODEL, hbase + n0 + nt * 16 + lr, k0, lh);
          hacc[0][nt] = wmma_bf16(a0, b, hacc[0][nt]);
          hacc[1][nt] = wmma_bf16(a1, b, hacc[1][nt]);
        }
      }
#pragma unroll
      for (int mt = 0; mt < 2; ++mt)
#pragma unroll
        for (int nt = 0; nt < 4; ++nt)
#pragma unroll
          for (int r = 0; r < 8; ++r) {
            int mm = mt * 16 + r + 8 * lh;
            int cc = n0 + nt * 16 + lr;
            Hd[mm * HP + cc] = f2bf(gelu_tanh(hacc[mt][nt][r]));
          }
    }
    __syncthreads();
    int d0 = wave * 128;
#pragma unroll 1
    for (int kt = 0; kt < 64; ++kt) {
      int k0 = kt * 32;
      Frag a0, a1;
      loadA(a0, Hd, HP, 0, k0, lr, lh);
      loadA(a1, Hd, HP, 16, k0, lr, lh);
#pragma unroll
      for (int nt = 0; nt < 8; ++nt) {
        Frag b;
        loadB(b, svb, HIDDEN, d0 + nt * 16 + lr, hbase + k0, lh);
        oacc[0][nt] = wmma_bf16(a0, b, oacc[0][nt]);
        oacc[1][nt] = wmma_bf16(a1, b, oacc[1][nt]);
      }
    }
    __syncthreads();
  }

  // epilogue: out = shared_out + sum_k gate_k * expert_out[g, e_k, c_k, :]
#pragma unroll
  for (int mt = 0; mt < 2; ++mt)
#pragma unroll
    for (int nt = 0; nt < 8; ++nt)
#pragma unroll
      for (int r = 0; r < 8; ++r) {
        int mm = mt * 16 + r + 8 * lh;
        int dd = wave * 128 + nt * 16 + lr;
        float v = oacc[mt][nt][r];
#pragma unroll
        for (int k = 0; k < 2; ++k) {
          int c = mC[mm * 2 + k];
          if (c >= 0) {
            int ee = mE[mm * 2 + k];
            v += mG[mm * 2 + k] *
                 bf2f(eo[(long long)((g * NEXP + ee) * NSLOT + c) * D_MODEL + dd]);
          }
        }
        out[(row0 + mm) * D_MODEL + dd] = v;
      }
}

// ---------------- host ----------------
extern "C" void kernel_launch(void* const* d_in, const int* in_sizes, int n_in,
                              void* d_out, int out_size, void* d_ws, size_t ws_size,
                              hipStream_t stream) {
  const float* x      = (const float*)d_in[0];
  const float* gw     = (const float*)d_in[1];
  const float* gb     = (const float*)d_in[2];
  const float* keys   = (const float*)d_in[3];
  const float* values = (const float*)d_in[4];
  const float* skw    = (const float*)d_in[5];
  const float* svw    = (const float*)d_in[6];
  float* out = (float*)d_out;

  char* ws = (char*)d_ws;
  size_t off = 0;
  auto take = [&](size_t bytes) {
    char* p = ws + off;
    off = (off + bytes + 255) & ~(size_t)255;
    return p;
  };
  unsigned short* xb  = (unsigned short*)take((size_t)NTOK * D_MODEL * 2);
  unsigned short* kb  = (unsigned short*)take((size_t)NEXP * HIDDEN * D_MODEL * 2);
  unsigned short* vb  = (unsigned short*)take((size_t)NEXP * D_MODEL * HIDDEN * 2);
  unsigned short* skb = (unsigned short*)take((size_t)HIDDEN * D_MODEL * 2);
  unsigned short* svb = (unsigned short*)take((size_t)D_MODEL * HIDDEN * 2);
  unsigned short* eo  = (unsigned short*)take((size_t)NGROUPS * NEXP * NSLOT * D_MODEL * 2);
  int*   disp = (int*)take((size_t)NGROUPS * NEXP * NSLOT * 2 * 4);
  int*   tokE = (int*)take((size_t)NGROUPS * GROUP * 2 * 4);
  int*   tokC = (int*)take((size_t)NGROUPS * GROUP * 2 * 4);
  float* tokG = (float*)take((size_t)NGROUPS * GROUP * 2 * 4);

  // 1) precision conversion (+ weight transposes for WMMA-friendly B loads)
  long long n2x = (long long)NTOK * D_MODEL / 2;
  convx_kernel<<<(int)((n2x + 255) / 256), 256, 0, stream>>>(x, (unsigned int*)xb, n2x);
  trans_kernel<<<dim3(HIDDEN / 32, D_MODEL / 32, NEXP), 256, 0, stream>>>(keys, kb, D_MODEL, HIDDEN);
  trans_kernel<<<dim3(D_MODEL / 32, HIDDEN / 32, NEXP), 256, 0, stream>>>(values, vb, HIDDEN, D_MODEL);
  trans_kernel<<<dim3(HIDDEN / 32, D_MODEL / 32, 1), 256, 0, stream>>>(skw, skb, D_MODEL, HIDDEN);
  trans_kernel<<<dim3(D_MODEL / 32, HIDDEN / 32, 1), 256, 0, stream>>>(svw, svb, HIDDEN, D_MODEL);

  // 2) gating / routing
  gate_kernel<<<NGROUPS, 128, 0, stream>>>(x, gw, gb, disp, tokE, tokC, tokG);

  // 3) routed experts: one block per (expert, group), fused 2-GEMM FFN
  size_t lds_e = (size_t)32 * AP * 2 + (size_t)32 * HP * 2 + 64 * 4;
  expert_kernel<<<NEXP * NGROUPS, 256, lds_e, stream>>>(xb, kb, vb, disp, eo);

  // 4) shared expert + combine
  size_t lds_s = (size_t)32 * AP * 2 + (size_t)32 * HP * 2 + 64 * 4 * 3;
  shared_kernel<<<NTOK / 32, 256, lds_s, stream>>>(xb, skb, svb, tokE, tokC, tokG, eo, out);
}
